// NodeAggregationModel_15195594293948
// MI455X (gfx1250) — matrix operand order repaired and checked by reading, not compiled, not run on hardware
//
#include <hip/hip_runtime.h>
#include <hip/hip_bf16.h>

#define NN     4096
#define F_IN   512
#define H1D    128
#define NHEADS 4
#define H2D    256
#define H3D    64
#define ALPHAV 0.2f
#define NEGV   -9.0e15f
#define KTILE  32
#define BPAD   40   // bf16 per column slot in LDS: 32 data + 8 pad (80B, kills bank conflicts)

typedef __bf16 bhalf;
typedef __bf16 v16bf __attribute__((ext_vector_type(16)));
typedef float  v8f   __attribute__((ext_vector_type(8)));

// ---------------- WMMA fragment loaders (CDNA5 wave32 layouts, ISA 7.12.2) ----
// A matrix 16x32 bf16, row-major source (lda = row stride in elements).
__device__ __forceinline__ v16bf load_a_frag(const bhalf* A, int lda, int r0, int k0, int lane) {
  const int row = r0 + (lane & 15);
  const int kh  = (lane >> 4) * 8;
  const bhalf* p = A + (size_t)row * lda + k0 + kh;
  v16bf a;
  reinterpret_cast<uint4*>(&a)[0] = reinterpret_cast<const uint4*>(p)[0];       // K = kh..kh+7
  reinterpret_cast<uint4*>(&a)[1] = reinterpret_cast<const uint4*>(p + 16)[0];  // K = 16+kh..23+kh
  return a;
}
// B matrix 32x16 bf16 from LDS tile stored K-major per column with BPAD stride.
__device__ __forceinline__ v16bf load_b_frag_lds(const bhalf* lb, int c0, int lane) {
  const bhalf* p = lb + (c0 + (lane & 15)) * BPAD + (lane >> 4) * 8;
  v16bf b;
  reinterpret_cast<uint4*>(&b)[0] = reinterpret_cast<const uint4*>(p)[0];       // K = kh..kh+7
  reinterpret_cast<uint4*>(&b)[1] = reinterpret_cast<const uint4*>(p + 16)[0];  // K = 16+kh..23+kh
  return b;
}

// ---------------- generic bf16 WMMA GEMM: C = act(A @ BT^T) ------------------
// Block = 8 waves = 128 rows x 64 cols. B tile (64x32) staged in LDS, double
// buffered, shared by all 8 waves. Software-pipelined: A fragment and next B
// tile are loaded one step ahead so WMMAs never drain the prefetch stream.
__global__ void __launch_bounds__(256)
gemm_bf16_wmma(const bhalf* __restrict__ A, const bhalf* __restrict__ BT,
               float* __restrict__ Cf, bhalf* __restrict__ Cb, bhalf* __restrict__ CbT,
               int M, int N, int K, int lda, int ldbt, int ldc, int ldct, int relu)
{
  __shared__ bhalf lbs[2][64 * BPAD];
  const int lane = threadIdx.x & 31;
  const int wave = threadIdx.x >> 5;
  const int r0 = blockIdx.x * 128 + wave * 16;
  const int c0 = blockIdx.y * 64;

  // cooperative B-tile fill: 256 threads x 16B = 64 cols x 32 K x bf16
  const int fcol = threadIdx.x >> 2;        // 0..63
  const int fchk = (threadIdx.x & 3) * 8;   // 0,8,16,24
  const bhalf* fsrc = BT + (size_t)(c0 + fcol) * ldbt + fchk;
  const int foff = fcol * BPAD + fchk;

  reinterpret_cast<uint4*>(&lbs[0][foff])[0] = reinterpret_cast<const uint4*>(fsrc)[0];
  v16bf a = load_a_frag(A, lda, r0, 0, lane);
  __syncthreads();

  v8f acc[4];
#pragma unroll
  for (int ct = 0; ct < 4; ++ct)
#pragma unroll
    for (int i = 0; i < 8; ++i) acc[ct][i] = 0.0f;

  const int nk = K / KTILE;
  int cur = 0;
  for (int t = 0; t < nk; ++t) {
    const bool more = (t + 1 < nk);
    uint4 nxtB;
    v16bf nxtA;
    if (more) {
      // issue next-tile loads first; WMMAs below only need loadcnt<=2
      nxtB = reinterpret_cast<const uint4*>(fsrc + (t + 1) * KTILE)[0];
      nxtA = load_a_frag(A, lda, r0, (t + 1) * KTILE, lane);
      __builtin_prefetch(A + (size_t)(r0 + (lane & 15)) * lda + (t + 2) * KTILE, 0, 1);
    }
    const bhalf* lb = lbs[cur];
    v16bf b0 = load_b_frag_lds(lb, 0, lane);
    v16bf b1 = load_b_frag_lds(lb, 16, lane);
    v16bf b2 = load_b_frag_lds(lb, 32, lane);
    v16bf b3 = load_b_frag_lds(lb, 48, lane);
    acc[0] = __builtin_amdgcn_wmma_f32_16x16x32_bf16(false, a, false, b0, (short)0, acc[0], false, false);
    acc[1] = __builtin_amdgcn_wmma_f32_16x16x32_bf16(false, a, false, b1, (short)0, acc[1], false, false);
    acc[2] = __builtin_amdgcn_wmma_f32_16x16x32_bf16(false, a, false, b2, (short)0, acc[2], false, false);
    acc[3] = __builtin_amdgcn_wmma_f32_16x16x32_bf16(false, a, false, b3, (short)0, acc[3], false, false);
    if (more) {
      reinterpret_cast<uint4*>(&lbs[cur ^ 1][foff])[0] = nxtB;  // ds_store (addrspace kept)
      __syncthreads();
      cur ^= 1;
      a = nxtA;
    }
  }

#pragma unroll
  for (int ct = 0; ct < 4; ++ct) {
#pragma unroll
    for (int v = 0; v < 8; ++v) {
      const int row = r0 + v + 8 * (lane >> 4);
      const int col = c0 + ct * 16 + (lane & 15);
      float val = acc[ct][v];
      if (relu) val = fmaxf(val, 0.0f);
      if (Cf)  Cf[(size_t)row * ldc + col]  = val;
      if (Cb)  Cb[(size_t)row * ldc + col]  = (bhalf)val;
      if (CbT) CbT[(size_t)col * ldct + row] = (bhalf)val;
    }
  }
}

// ---------------- prep: f32 -> bf16 copy / transpose-copy --------------------
__global__ void __launch_bounds__(256)
cvt_f32_bf16(const float* __restrict__ in, bhalf* __restrict__ out, int n)
{
  int i = blockIdx.x * 256 + threadIdx.x;
  if (i < n) out[i] = (bhalf)in[i];
}

__global__ void __launch_bounds__(256)
transpose_cvt(const float* __restrict__ in, bhalf* __restrict__ out,
              int R, int C, size_t inBatch, size_t outBatch)
{
  const float* ib = in + blockIdx.z * inBatch;
  bhalf* ob = out + blockIdx.z * outBatch;
  int i = blockIdx.x * 256 + threadIdx.x;
  if (i < R * C) {
    int r = i / C, c = i - r * C;
    ob[(size_t)c * R + r] = (bhalf)ib[i];
  }
}

// ---------------- s1/s2 = Wh . a1 / a2 (wave per (h,row)) --------------------
__global__ void __launch_bounds__(256)
compute_scores(const float* __restrict__ Wh, const float* __restrict__ a,
               float* __restrict__ s1, float* __restrict__ s2)
{
  const int lane = threadIdx.x & 31, wave = threadIdx.x >> 5;
  const int h = blockIdx.y;
  const int row = blockIdx.x * 8 + wave;
  const float* wr = Wh + ((size_t)h * NN + row) * H1D;
  const float* a1 = a + h * 2 * H1D;
  const float* a2 = a1 + H1D;
  float x1 = 0.0f, x2 = 0.0f;
  for (int c = lane; c < H1D; c += 32) { float w = wr[c]; x1 += w * a1[c]; x2 += w * a2[c]; }
#pragma unroll
  for (int o = 16; o >= 1; o >>= 1) { x1 += __shfl_xor(x1, o, 32); x2 += __shfl_xor(x2, o, 32); }
  if (lane == 0) { s1[h * NN + row] = x1; s2[h * NN + row] = x2; }
}

// ---------------- per-row softmax stats over masked e[n,m] -------------------
__global__ void __launch_bounds__(256)
row_stats(const float* __restrict__ adj, const float* __restrict__ s1,
          const float* __restrict__ s2, float* __restrict__ rmax, float* __restrict__ rsum)
{
  const int lane = threadIdx.x & 31, wave = threadIdx.x >> 5;
  const int h = blockIdx.y;
  const int row = blockIdx.x * 8 + wave;
  const float s1r = s1[h * NN + row];
  const float* arow = adj + (size_t)row * NN;
  const float* s2h = s2 + h * NN;
  float mx = -3.0e38f, sm = 0.0f;
  for (int m = lane; m < NN; m += 32) {
    __builtin_prefetch(arow + m + 512, 0, 1);
    float x = s1r + s2h[m];
    x = x > 0.0f ? x : ALPHAV * x;
    float v = arow[m] > 0.0f ? x : NEGV;
    if (v > mx) { sm = sm * __expf(mx - v) + 1.0f; mx = v; }
    else        { sm += __expf(v - mx); }
  }
#pragma unroll
  for (int o = 16; o >= 1; o >>= 1) {
    float mo = __shfl_xor(mx, o, 32), so = __shfl_xor(sm, o, 32);
    float M = fmaxf(mx, mo);
    sm = sm * __expf(mx - M) + so * __expf(mo - M);
    mx = M;
  }
  if (lane == 0) { rmax[h * NN + row] = mx; rsum[h * NN + row] = sm; }
}

// ---------------- WMMA attention: xr[n, h*128+c] = relu(sum_m att*Wh) --------
// Block = 8 waves x 16 rows (one head). Probability tiles built directly in
// A-fragment layout; WhT tile (128 x 32 bf16) staged in LDS, double buffered,
// shared by all 8 waves of the block.
__global__ void __launch_bounds__(256)
gat_attention_wmma(const float* __restrict__ adj, const bhalf* __restrict__ WhTb,
                   const float* __restrict__ s1, const float* __restrict__ s2,
                   const float* __restrict__ rmax, const float* __restrict__ rsum,
                   bhalf* __restrict__ xrb)
{
  __shared__ bhalf lws[2][H1D * BPAD];
  const int lane = threadIdx.x & 31;
  const int wave = threadIdx.x >> 5;
  const int h = blockIdx.y;
  const int r0 = (blockIdx.x * 8 + wave) * 16;
  const int row = r0 + (lane & 15);
  const int kh = (lane >> 4) * 8;
  const float s1r = s1[h * NN + row];
  const float mr  = rmax[h * NN + row];
  const float ir  = 1.0f / rsum[h * NN + row];
  const float* s2h = s2 + h * NN;
  const bhalf* whT = WhTb + (size_t)h * H1D * NN;

  // cooperative WhT tile fill: 128 cols x 32 K -> 512 x 16B chunks, 2/thread
  const int u0 = threadIdx.x, u1 = threadIdx.x + 256;
  const int fc0 = u0 >> 2, fk0 = (u0 & 3) * 8;
  const int fc1 = u1 >> 2, fk1 = (u1 & 3) * 8;
  const bhalf* fsrc0 = whT + (size_t)fc0 * NN + fk0;
  const bhalf* fsrc1 = whT + (size_t)fc1 * NN + fk1;
  const int foff0 = fc0 * BPAD + fk0;
  const int foff1 = fc1 * BPAD + fk1;

  reinterpret_cast<uint4*>(&lws[0][foff0])[0] = reinterpret_cast<const uint4*>(fsrc0)[0];
  reinterpret_cast<uint4*>(&lws[0][foff1])[0] = reinterpret_cast<const uint4*>(fsrc1)[0];
  __syncthreads();

  v8f acc[8];
#pragma unroll
  for (int ct = 0; ct < 8; ++ct)
#pragma unroll
    for (int i = 0; i < 8; ++i) acc[ct][i] = 0.0f;

  int cur = 0;
  for (int mt = 0; mt < NN; mt += KTILE) {
    const bool more = (mt + KTILE < NN);
    // adj / s2 loads for this step first ...
    const float* ap = adj + (size_t)row * NN + mt + kh;
    const float* sp = s2h + mt + kh;
    const float4 A0 = reinterpret_cast<const float4*>(ap)[0];
    const float4 A1 = reinterpret_cast<const float4*>(ap + 4)[0];
    const float4 A2 = reinterpret_cast<const float4*>(ap + 16)[0];
    const float4 A3 = reinterpret_cast<const float4*>(ap + 20)[0];
    const float4 S0 = reinterpret_cast<const float4*>(sp)[0];
    const float4 S1 = reinterpret_cast<const float4*>(sp + 4)[0];
    const float4 S2 = reinterpret_cast<const float4*>(sp + 16)[0];
    const float4 S3 = reinterpret_cast<const float4*>(sp + 20)[0];
    __builtin_prefetch(ap + 256, 0, 1);
    // ... then next WhT tile loads, so they stay in flight through the VALU+WMMA work
    uint4 nxt0, nxt1;
    if (more) {
      nxt0 = reinterpret_cast<const uint4*>(fsrc0 + mt + KTILE)[0];
      nxt1 = reinterpret_cast<const uint4*>(fsrc1 + mt + KTILE)[0];
    }
    v16bf p;
    {
      float av[16] = {A0.x, A0.y, A0.z, A0.w, A1.x, A1.y, A1.z, A1.w,
                      A2.x, A2.y, A2.z, A2.w, A3.x, A3.y, A3.z, A3.w};
      float sv[16] = {S0.x, S0.y, S0.z, S0.w, S1.x, S1.y, S1.z, S1.w,
                      S2.x, S2.y, S2.z, S2.w, S3.x, S3.y, S3.z, S3.w};
#pragma unroll
      for (int e = 0; e < 16; ++e) {
        float x = s1r + sv[e];
        x = x > 0.0f ? x : ALPHAV * x;
        float pv = av[e] > 0.0f ? __expf(x - mr) * ir : 0.0f;
        p[e] = (bhalf)pv;
      }
    }
    const bhalf* lb = lws[cur];
    {
      v16bf b0 = load_b_frag_lds(lb, 0, lane);
      v16bf b1 = load_b_frag_lds(lb, 16, lane);
      v16bf b2 = load_b_frag_lds(lb, 32, lane);
      v16bf b3 = load_b_frag_lds(lb, 48, lane);
      acc[0] = __builtin_amdgcn_wmma_f32_16x16x32_bf16(false, p, false, b0, (short)0, acc[0], false, false);
      acc[1] = __builtin_amdgcn_wmma_f32_16x16x32_bf16(false, p, false, b1, (short)0, acc[1], false, false);
      acc[2] = __builtin_amdgcn_wmma_f32_16x16x32_bf16(false, p, false, b2, (short)0, acc[2], false, false);
      acc[3] = __builtin_amdgcn_wmma_f32_16x16x32_bf16(false, p, false, b3, (short)0, acc[3], false, false);
    }
    {
      v16bf b4 = load_b_frag_lds(lb, 64, lane);
      v16bf b5 = load_b_frag_lds(lb, 80, lane);
      v16bf b6 = load_b_frag_lds(lb, 96, lane);
      v16bf b7 = load_b_frag_lds(lb, 112, lane);
      acc[4] = __builtin_amdgcn_wmma_f32_16x16x32_bf16(false, p, false, b4, (short)0, acc[4], false, false);
      acc[5] = __builtin_amdgcn_wmma_f32_16x16x32_bf16(false, p, false, b5, (short)0, acc[5], false, false);
      acc[6] = __builtin_amdgcn_wmma_f32_16x16x32_bf16(false, p, false, b6, (short)0, acc[6], false, false);
      acc[7] = __builtin_amdgcn_wmma_f32_16x16x32_bf16(false, p, false, b7, (short)0, acc[7], false, false);
    }
    if (more) {
      reinterpret_cast<uint4*>(&lws[cur ^ 1][foff0])[0] = nxt0;  // ds_store
      reinterpret_cast<uint4*>(&lws[cur ^ 1][foff1])[0] = nxt1;  // ds_store
      __syncthreads();
      cur ^= 1;
    }
  }

#pragma unroll
  for (int ct = 0; ct < 8; ++ct) {
#pragma unroll
    for (int v = 0; v < 8; ++v) {
      const int orow = r0 + v + 8 * (lane >> 4);
      const int ocol = h * H1D + ct * 16 + (lane & 15);
      xrb[(size_t)orow * (NHEADS * H1D) + ocol] = (bhalf)fmaxf(acc[ct][v], 0.0f);
    }
  }
}

// ---------------- final row softmax over g2 (64 cols) ------------------------
__global__ void __launch_bounds__(256)
softmax_rows(const float* __restrict__ g2, float* __restrict__ out)
{
  const int lane = threadIdx.x & 31, wave = threadIdx.x >> 5;
  const int row = blockIdx.x * 8 + wave;
  const float* r = g2 + (size_t)row * H3D;
  float a = r[lane], b = r[lane + 32];
  float mx = fmaxf(a, b);
#pragma unroll
  for (int o = 16; o >= 1; o >>= 1) mx = fmaxf(mx, __shfl_xor(mx, o, 32));
  float ea = __expf(a - mx), eb = __expf(b - mx);
  float s = ea + eb;
#pragma unroll
  for (int o = 16; o >= 1; o >>= 1) s += __shfl_xor(s, o, 32);
  float inv = 1.0f / s;
  out[(size_t)row * H3D + lane] = ea * inv;
  out[(size_t)row * H3D + lane + 32] = eb * inv;
}

// ---------------- host orchestration -----------------------------------------
extern "C" void kernel_launch(void* const* d_in, const int* in_sizes, int n_in,
                              void* d_out, int out_size, void* d_ws, size_t ws_size,
                              hipStream_t stream) {
  const float* x    = (const float*)d_in[0];  // 4096x512
  const float* adj  = (const float*)d_in[1];  // 4096x4096
  const float* Watt = (const float*)d_in[2];  // 4x512x128
  const float* avec = (const float*)d_in[3];  // 4x256
  const float* W1   = (const float*)d_in[4];  // 512x256
  const float* W2   = (const float*)d_in[5];  // 256x64
  float* out = (float*)d_out;                 // [g2 | softmax(g2)], each 4096x64

  char* ws = (char*)d_ws;
  size_t off = 0;
  auto alloc = [&](size_t bytes) { char* p = ws + off; off += (bytes + 255) & ~(size_t)255; return p; };
  bhalf* xb     = (bhalf*)alloc((size_t)NN * F_IN * 2);
  bhalf* adjb   = (bhalf*)alloc((size_t)NN * NN * 2);
  bhalf* WattTb = (bhalf*)alloc((size_t)NHEADS * H1D * F_IN * 2);
  bhalf* W1Tb   = (bhalf*)alloc((size_t)H2D * (NHEADS * H1D) * 2);
  bhalf* W2Tb   = (bhalf*)alloc((size_t)H3D * H2D * 2);
  float* Wh     = (float*)alloc((size_t)NHEADS * NN * H1D * 4);
  bhalf* WhTb   = (bhalf*)alloc((size_t)NHEADS * H1D * NN * 2);
  float* s1     = (float*)alloc((size_t)NHEADS * NN * 4);
  float* s2     = (float*)alloc((size_t)NHEADS * NN * 4);
  float* rmax   = (float*)alloc((size_t)NHEADS * NN * 4);
  float* rsum   = (float*)alloc((size_t)NHEADS * NN * 4);
  bhalf* xrb    = (bhalf*)alloc((size_t)NN * (NHEADS * H1D) * 2);
  bhalf* PTb    = (bhalf*)alloc((size_t)H2D * NN * 2);
  bhalf* g1b    = (bhalf*)alloc((size_t)NN * H2D * 2);
  bhalf* QTb    = (bhalf*)alloc((size_t)H3D * NN * 2);

  // 1) bf16 copies
  cvt_f32_bf16<<<(NN * F_IN + 255) / 256, 256, 0, stream>>>(x, xb, NN * F_IN);
  cvt_f32_bf16<<<((size_t)NN * NN + 255) / 256, 256, 0, stream>>>(adj, adjb, NN * NN);
  // weights transposed to (N x K) bf16
  transpose_cvt<<<dim3((F_IN * H1D + 255) / 256, 1, NHEADS), 256, 0, stream>>>(
      Watt, WattTb, F_IN, H1D, (size_t)F_IN * H1D, (size_t)H1D * F_IN);
  transpose_cvt<<<dim3(((NHEADS * H1D) * H2D + 255) / 256, 1, 1), 256, 0, stream>>>(
      W1, W1Tb, NHEADS * H1D, H2D, 0, 0);
  transpose_cvt<<<dim3((H2D * H3D + 255) / 256, 1, 1), 256, 0, stream>>>(
      W2, W2Tb, H2D, H3D, 0, 0);

  // 2) Wh[h] = x @ W_att[h]  (also emit WhT bf16 for attention B-fragments)
  for (int h = 0; h < NHEADS; ++h) {
    gemm_bf16_wmma<<<dim3(NN / 128, H1D / 64), 256, 0, stream>>>(
        xb, WattTb + (size_t)h * H1D * F_IN,
        Wh + (size_t)h * NN * H1D, nullptr, WhTb + (size_t)h * H1D * NN,
        NN, H1D, F_IN, F_IN, F_IN, H1D, NN, 0);
  }

  // 3) attention scores + row softmax stats
  compute_scores<<<dim3(NN / 8, NHEADS), 256, 0, stream>>>(Wh, avec, s1, s2);
  row_stats<<<dim3(NN / 8, NHEADS), 256, 0, stream>>>(adj, s1, s2, rmax, rsum);

  // 4) flash-style WMMA attention -> xr (relu'd concat, bf16 row-major)
  gat_attention_wmma<<<dim3(NN / 128, NHEADS), 256, 0, stream>>>(
      adj, WhTb, s1, s2, rmax, rsum, xrb);

  // 5) P = xr @ W1 -> PT (bf16, K-major for next GEMM)
  gemm_bf16_wmma<<<dim3(NN / 128, H2D / 64), 256, 0, stream>>>(
      xrb, W1Tb, nullptr, nullptr, PTb,
      NN, H2D, NHEADS * H1D, NHEADS * H1D, NHEADS * H1D, H2D, NN, 0);

  // 6) g1 = relu(adj @ P) -> bf16 row-major
  gemm_bf16_wmma<<<dim3(NN / 128, H2D / 64), 256, 0, stream>>>(
      adjb, PTb, nullptr, g1b, nullptr,
      NN, H2D, NN, NN, NN, H2D, NN, 1);

  // 7) Q = g1 @ W2 -> QT (bf16)
  gemm_bf16_wmma<<<dim3(NN / 128, H3D / 64), 256, 0, stream>>>(
      g1b, W2Tb, nullptr, nullptr, QTb,
      NN, H3D, H2D, H2D, H2D, H3D, NN, 0);

  // 8) g2 = adj @ Q -> f32 output (first half of d_out)
  gemm_bf16_wmma<<<dim3(NN / 128, H3D / 64), 256, 0, stream>>>(
      adjb, QTb, out, nullptr, nullptr,
      NN, H3D, NN, NN, NN, H3D, NN, 0);

  // 9) softmax(g2, axis=1) -> second half of d_out
  softmax_rows<<<NN / 8, 256, 0, stream>>>(out, out + (size_t)NN * H3D);
}
